// MultiScaleEquivariantResidualNet_14851996909783
// MI455X (gfx1250) — matrix operand order repaired and checked by reading, not compiled, not run on hardware
//
#include <hip/hip_runtime.h>
#include <hip/hip_bf16.h>

typedef __attribute__((ext_vector_type(16))) _Float16 v16h;
typedef __attribute__((ext_vector_type(8)))  _Float16 v8h;
typedef __attribute__((ext_vector_type(8)))  float    v8f;

#define HD     64
#define HEADS  8
#define NBAS   16
#define LAYERS 4
#define GRAPHS 200

__device__ __forceinline__ float sigmoidf_(float x) { return 1.0f / (1.0f + __expf(-x)); }

__global__ void fill_kernel(float* __restrict__ p, float v, int n) {
  int i = blockIdx.x * blockDim.x + threadIdx.x;
  if (i < n) p[i] = v;
}

// C[M,N] = act(A[M,K] @ B[K,N] + bias). ACT: 0=none 1=relu. bias non-null.
// REQUIRES M % 16 == 0 (true for all call sites: M is 240000 or 20000) and
// M*K, M*N < 2^31 (true: max 46.1M elements) so int32 offsets are safe.
// Block = 8 waves sharing one 64-column B strip staged in LDS as f16 (padded
// stride, zero-filled tail). Each wave computes a 16x64 strip of C: 4 n-tiles
// share one A fragment (b128 global loads); B fragments via 2x ds_load_b128.
template<int K, int ACT>
__global__ __launch_bounds__(256)
void gemm_wmma(const float* __restrict__ A, const float* __restrict__ B,
               const float* __restrict__ bias, float* __restrict__ C, int M, int N)
{
  constexpr int NT  = 4;
  constexpr int KCH = (K + 31) / 32;
  constexpr int KP  = KCH * 32 + 8;             // padded LDS stride (halves); 16B-aligned frags
  __shared__ _Float16 Bs[64 * KP];

  const int tid  = threadIdx.x;
  const int lane = tid & 31;
  const int wid  = tid >> 5;
  const int n0   = blockIdx.y << 6;

  // cooperative stage: B[k][n0+n] (f32, coalesced in n) -> Bs[n*KP + k] (f16)
  for (int idx = tid; idx < 64 * KP; idx += 256) {
    int k = idx >> 6, n = idx & 63;
    int kk = (k < K) ? k : K - 1;
    float vv = B[kk * N + n0 + n];
    Bs[n * KP + k] = (_Float16)((k < K) ? vv : 0.0f);
  }
  __syncthreads();

  const int tiles_m = M >> 4;
  const int mt = blockIdx.x * 8 + wid;
  if (mt >= tiles_m) return;                    // wave-uniform, after barrier
  const int m0  = mt << 4;
  const int l15 = lane & 15;
  const int hi  = lane >> 4;
  const int abase = (m0 + l15) * K;             // int32 offset: saddr+voffset addressing

  v8f acc[NT] = {};
#pragma unroll
  for (int c = 0; c < KCH; ++c) {
    const int k0 = c * 32;
    v16h a;
    if (k0 + 32 <= K) {                         // full chunk (compile-time)
      if (K % 4 == 0) {                         // 4x global_load_b128
        const float4 p0 = *(const float4*)(A + abase + k0 + 8 * hi);
        const float4 p1 = *(const float4*)(A + abase + k0 + 8 * hi + 4);
        const float4 p2 = *(const float4*)(A + abase + k0 + 16 + 8 * hi);
        const float4 p3 = *(const float4*)(A + abase + k0 + 16 + 8 * hi + 4);
        a[0]=(_Float16)p0.x; a[1]=(_Float16)p0.y; a[2]=(_Float16)p0.z; a[3]=(_Float16)p0.w;
        a[4]=(_Float16)p1.x; a[5]=(_Float16)p1.y; a[6]=(_Float16)p1.z; a[7]=(_Float16)p1.w;
        a[8]=(_Float16)p2.x; a[9]=(_Float16)p2.y; a[10]=(_Float16)p2.z; a[11]=(_Float16)p2.w;
        a[12]=(_Float16)p3.x; a[13]=(_Float16)p3.y; a[14]=(_Float16)p3.z; a[15]=(_Float16)p3.w;
      } else {
#pragma unroll
        for (int j = 0; j < 8; ++j) a[j]     = (_Float16)A[abase + k0 + 8 * hi + j];
#pragma unroll
        for (int j = 0; j < 8; ++j) a[8 + j] = (_Float16)A[abase + k0 + 16 + 8 * hi + j];
      }
    } else {                                    // partial chunk: clamp addr, select value
#pragma unroll
      for (int j = 0; j < 8; ++j) {
        int k = k0 + 8 * hi + j;
        int kk = (k < K) ? k : 0;
        float vv = A[abase + kk];
        a[j] = (_Float16)((k < K) ? vv : 0.0f);
      }
#pragma unroll
      for (int j = 0; j < 8; ++j) {
        int k = k0 + 16 + 8 * hi + j;
        int kk = (k < K) ? k : 0;
        float vv = A[abase + kk];
        a[8 + j] = (_Float16)((k < K) ? vv : 0.0f);
      }
    }
#pragma unroll
    for (int t = 0; t < NT; ++t) {
      // B fragment: lane holds column (t*16+l15), K = k0 + 8*hi + j and k0 + 16 + 8*hi + j
      const _Float16* bp = Bs + (t * 16 + l15) * KP + k0 + 8 * hi;
      v8h b0 = *(const v8h*)bp;                 // ds_load_b128
      v8h b1 = *(const v8h*)(bp + 16);          // ds_load_b128
      v16h b;
#pragma unroll
      for (int j = 0; j < 8; ++j) { b[j] = b0[j]; b[8 + j] = b1[j]; }
      acc[t] = __builtin_amdgcn_wmma_f32_16x16x32_f16(false, a, false, b, (short)0,
                                                      acc[t], false, false);
    }
  }

  // D rows are consecutive: M = m0 + 8*hi + r, N = n0 + t*16 + l15
#pragma unroll
  for (int t = 0; t < NT; ++t) {
    const int nc = n0 + t * 16 + l15;
    const float bv = bias[nc];
    const int cbase = (m0 + 8 * hi) * N + nc;   // int32 offset: saddr+voffset stores
#pragma unroll
    for (int r = 0; r < 8; ++r) {
      float v = acc[t][r] + bv;
      if (ACT == 1) v = fmaxf(v, 0.0f);
      C[cbase + r * N] = v;
    }
  }
}

__global__ void basis_kernel(const float* __restrict__ pos, const int* __restrict__ src,
                             const int* __restrict__ dst, float* __restrict__ basis, int E)
{
  int e = blockIdx.x * blockDim.x + threadIdx.x;
  if (e >= E) return;
  int s = src[e], d = dst[e];
  float dx = pos[d * 3 + 0] - pos[s * 3 + 0];
  float dy = pos[d * 3 + 1] - pos[s * 3 + 1];
  float dz = pos[d * 3 + 2] - pos[s * 3 + 2];
  float r  = sqrtf(dx * dx + dy * dy + dz * dz + 1e-12f);
#pragma unroll
  for (int i = 0; i < NBAS; ++i) {
    float c = (float)i * (5.0f / 15.0f);
    float t = (r - c) * (16.0f / 5.0f);
    basis[(size_t)e * NBAS + i] = __expf(-t * t);
  }
}

__global__ void msg_scatter_kernel(const float* __restrict__ hm, const float* __restrict__ ea,
                                   const float* __restrict__ we, const int* __restrict__ src,
                                   const int* __restrict__ dst, float* __restrict__ agg, int E)
{
  int t = blockIdx.x * blockDim.x + threadIdx.x;
  if (t >= E * HD) return;
  int e = t >> 6, c = t & 63;
  int s = src[e], d = dst[e];
  float v = hm[s * HD + c] * ea[(size_t)e * HD + c] * we[(size_t)e * HD + c]
          * 0.28867513459481287f;               // 1/sqrt(12)
  atomicAdd(&agg[d * HD + c], v);
}

__global__ void combine_kernel(const float* __restrict__ hs, const float* __restrict__ agg,
                               float* __restrict__ h, int N, int do_silu)
{
  int t = blockIdx.x * blockDim.x + threadIdx.x;
  if (t >= N * HD) return;
  float v = hs[t] + agg[t];
  if (do_silu) v = v * sigmoidf_(v);
  h[t] = v;
}

__device__ __forceinline__ void atomicMaxF(float* addr, float val) {
  int* ai = (int*)addr;
  int cur = *ai;
  while (__int_as_float(cur) < val) {
    int prev = atomicCAS(ai, cur, __float_as_int(val));
    if (prev == cur) break;
    cur = prev;
  }
}

__global__ void score_kernel(const float* __restrict__ q, const float* __restrict__ k,
                             const int* __restrict__ src, const int* __restrict__ dst,
                             float* __restrict__ sc, float* __restrict__ smax, int E)
{
  int t = blockIdx.x * blockDim.x + threadIdx.x;
  if (t >= E * HEADS) return;
  int e = t >> 3, hh = t & 7;
  int s = src[e], d = dst[e];
  const float4* qp = (const float4*)(q + (size_t)d * (HEADS * HD) + hh * HD);
  const float4* kp = (const float4*)(k + (size_t)s * (HEADS * HD) + hh * HD);
  float acc = 0.0f;
#pragma unroll
  for (int c = 0; c < HD / 4; ++c) {
    float4 a = qp[c], b = kp[c];
    acc += a.x * b.x + a.y * b.y + a.z * b.z + a.w * b.w;
  }
  acc *= 0.125f;                                // 1/sqrt(64)
  sc[t] = acc;
  atomicMaxF(&smax[d * HEADS + hh], acc);
}

__global__ void expden_kernel(float* __restrict__ sc, const float* __restrict__ smax,
                              const int* __restrict__ dst, float* __restrict__ den, int E)
{
  int t = blockIdx.x * blockDim.x + threadIdx.x;
  if (t >= E * HEADS) return;
  int e = t >> 3, hh = t & 7;
  int d = dst[e];
  float ex = __expf(sc[t] - smax[d * HEADS + hh]);
  sc[t] = ex;
  atomicAdd(&den[d * HEADS + hh], ex);
}

__global__ void attn_agg_kernel(const float* __restrict__ ex, const float* __restrict__ den,
                                const float* __restrict__ v, const int* __restrict__ src,
                                const int* __restrict__ dst, float* __restrict__ agg2, int E)
{
  int t = blockIdx.x * blockDim.x + threadIdx.x;
  if (t >= E * HD) return;
  int e = t >> 6, c = t & 63;
  int s = src[e], d = dst[e];
  float acc = 0.0f;
#pragma unroll
  for (int hh = 0; hh < HEADS; ++hh) {
    float alpha = ex[(size_t)e * HEADS + hh] / (den[d * HEADS + hh] + 1e-16f);
    acc += alpha * v[s * (HEADS * HD) + hh * HD + c];
  }
  atomicAdd(&agg2[d * HD + c], acc * (1.0f / HEADS));
}

__global__ void xlocal_kernel(const float* __restrict__ agg2, const float* __restrict__ hskip,
                              const float* __restrict__ h, const float* __restrict__ g,
                              const float* __restrict__ b, float* __restrict__ xc, int N)
{
  int n = blockIdx.x * blockDim.x + threadIdx.x;
  if (n >= N) return;
  float s = 0.0f, s2 = 0.0f;
  for (int c = 0; c < HD; ++c) {
    float t = agg2[n * HD + c] + hskip[n * HD + c] + h[n * HD + c];
    s += t; s2 += t * t;
  }
  float m  = s / HD;
  float is = rsqrtf(s2 / HD - m * m + 1e-5f);
  for (int c = 0; c < HD; ++c) {
    float t = agg2[n * HD + c] + hskip[n * HD + c] + h[n * HD + c];
    xc[n * (3 * HD) + c] = (t - m) * is * g[c] + b[c];
  }
}

__global__ void ga_kernel(const float* __restrict__ h, const float* __restrict__ Wga,
                          const float* __restrict__ bga, const int* __restrict__ batch,
                          float* __restrict__ num, float* __restrict__ cnt, int N)
{
  int n = blockIdx.x * blockDim.x + threadIdx.x;
  if (n >= N) return;
  float s = bga[0];
  for (int c = 0; c < HD; ++c) s += h[n * HD + c] * Wga[c];
  float aw = sigmoidf_(s);
  int g = batch[n];
  for (int c = 0; c < HD; ++c) atomicAdd(&num[g * HD + c], h[n * HD + c] * aw);
  atomicAdd(&cnt[g], 1.0f);
}

__global__ void xglobal_kernel(const float* __restrict__ h, const float* __restrict__ num,
                               const float* __restrict__ cnt, const int* __restrict__ batch,
                               const float* __restrict__ g, const float* __restrict__ b,
                               float* __restrict__ xc, int N)
{
  int n = blockIdx.x * blockDim.x + threadIdx.x;
  if (n >= N) return;
  int gi = batch[n];
  float ic = 1.0f / fmaxf(cnt[gi], 1.0f);
  float s = 0.0f, s2 = 0.0f;
  for (int c = 0; c < HD; ++c) { float a = h[n * HD + c];          s += a; s2 += a * a; }
  for (int c = 0; c < HD; ++c) { float a = num[gi * HD + c] * ic;  s += a; s2 += a * a; }
  float m  = s / (2 * HD);
  float is = rsqrtf(s2 / (2 * HD) - m * m + 1e-5f);
  for (int c = 0; c < HD; ++c)
    xc[n * (3 * HD) + HD + c]     = (h[n * HD + c] - m)         * is * g[c]      + b[c];
  for (int c = 0; c < HD; ++c)
    xc[n * (3 * HD) + 2 * HD + c] = (num[gi * HD + c] * ic - m) * is * g[HD + c] + b[HD + c];
}

__global__ void classifier_kernel(const float* __restrict__ proj, const float* __restrict__ Wc,
                                  const float* __restrict__ bc, float* __restrict__ out, int N)
{
  int n = blockIdx.x * blockDim.x + threadIdx.x;
  if (n >= N) return;
  const float4* pp = (const float4*)(proj + (size_t)n * 128);
  const float4* wp = (const float4*)Wc;
  float s = bc[0];
#pragma unroll
  for (int j = 0; j < 32; ++j) {
    float4 a = pp[j], b = wp[j];
    s += a.x * b.x + a.y * b.y + a.z * b.z + a.w * b.w;
  }
  out[n] = s;
}

extern "C" void kernel_launch(void* const* d_in, const int* in_sizes, int n_in,
                              void* d_out, int out_size, void* d_ws, size_t ws_size,
                              hipStream_t stream)
{
  (void)n_in; (void)out_size; (void)ws_size;
  const float* pos       = (const float*)d_in[0];
  const float* x         = (const float*)d_in[1];
  const float* edge_attr = (const float*)d_in[2];
  const int*   edge_index= (const int*)d_in[3];
  const int*   batch     = (const int*)d_in[4];
  const float* W_ep = (const float*)d_in[5];   const float* b_ep = (const float*)d_in[6];
  const float* Wm0  = (const float*)d_in[7];   const float* Ws0  = (const float*)d_in[8];
  const float* R10  = (const float*)d_in[9];   const float* R20  = (const float*)d_in[10];
  const float* Wm   = (const float*)d_in[11];  const float* Ws   = (const float*)d_in[12];
  const float* R1   = (const float*)d_in[13];  const float* R2   = (const float*)d_in[14];
  const float* Wq   = (const float*)d_in[15];  const float* Wk   = (const float*)d_in[16];
  const float* Wv   = (const float*)d_in[17];  const float* Wskip= (const float*)d_in[18];
  const float* b_skip=(const float*)d_in[19];
  const float* g_ln_l=(const float*)d_in[20];  const float* b_ln_l=(const float*)d_in[21];
  const float* W_ga = (const float*)d_in[22];  const float* b_ga = (const float*)d_in[23];
  const float* g_ln_g=(const float*)d_in[24];  const float* b_ln_g=(const float*)d_in[25];
  const float* P1   = (const float*)d_in[26];  const float* P2   = (const float*)d_in[27];
  const float* Wc   = (const float*)d_in[28];  const float* bc   = (const float*)d_in[29];

  const int N = in_sizes[0] / 3;
  const int E = in_sizes[2] / 19;
  const int* src = edge_index;
  const int* dst = edge_index + E;

  // workspace layout (floats), with post-conv reuse: t1->q, we->k, ea->v, basis->score
  float* ws = (float*)d_ws;
  size_t off = 0;
  auto alloc = [&](size_t n) { float* p = ws + off; off += (n + 63) & ~(size_t)63; return p; };
  float* zeros = alloc(512);                // zero bias for bias-less GEMMs
  float* basis = alloc((size_t)E * NBAS);
  float* ea    = alloc((size_t)E * HD);
  float* t1    = alloc((size_t)E * HD);
  float* we    = alloc((size_t)E * HD);
  float* hbuf  = alloc((size_t)N * HD);
  float* hm    = alloc((size_t)N * HD);     // also hskip
  float* hs    = alloc((size_t)N * HD);
  float* agg   = alloc((size_t)N * HD);     // also attention agg
  float* smax  = alloc((size_t)N * HEADS);
  float* den   = alloc((size_t)N * HEADS);
  float* xc    = alloc((size_t)N * 3 * HD);
  float* thid  = alloc((size_t)N * HD);
  float* num   = alloc((size_t)GRAPHS * HD);
  float* cnt   = alloc((size_t)GRAPHS);
  float* q = t1; float* kb = we; float* v = ea; float* sc = basis;

  auto gemm = [&](const float* A, const float* B, const float* bias, float* C,
                  int M, int Nn, int K, int act) {
    const float* bp = bias ? bias : zeros;
    dim3 grid((unsigned)(((M >> 4) + 7) >> 3), (unsigned)(Nn >> 6));
    if (act == 0) {
      switch (K) {
        case 16:  gemm_wmma<16, 0><<<grid, 256, 0, stream>>>(A, B, bp, C, M, Nn); break;
        case 19:  gemm_wmma<19, 0><<<grid, 256, 0, stream>>>(A, B, bp, C, M, Nn); break;
        case 23:  gemm_wmma<23, 0><<<grid, 256, 0, stream>>>(A, B, bp, C, M, Nn); break;
        case 64:  gemm_wmma<64, 0><<<grid, 256, 0, stream>>>(A, B, bp, C, M, Nn); break;
        case 192: gemm_wmma<192,0><<<grid, 256, 0, stream>>>(A, B, bp, C, M, Nn); break;
      }
    } else {
      switch (K) {
        case 16:  gemm_wmma<16, 1><<<grid, 256, 0, stream>>>(A, B, bp, C, M, Nn); break;
        case 192: gemm_wmma<192,1><<<grid, 256, 0, stream>>>(A, B, bp, C, M, Nn); break;
      }
    }
  };
  auto fillf = [&](float* p, float val, size_t n) {
    fill_kernel<<<(unsigned)((n + 255) >> 8), 256, 0, stream>>>(p, val, (int)n);
  };

  fillf(zeros, 0.0f, 512);

  // edge basis + edge feature projection
  basis_kernel<<<(E + 255) / 256, 256, 0, stream>>>(pos, src, dst, basis, E);
  gemm(edge_attr, W_ep, b_ep, ea, E, HD, 19, 0);

  // equivariant conv stack
  for (int l = 0; l < LAYERS; ++l) {
    const float* Ah  = (l == 0) ? x   : hbuf;
    const int    Kh  = (l == 0) ? 23  : HD;
    const float* R1l = (l == 0) ? R10 : R1 + (size_t)(l - 1) * NBAS * HD;
    const float* R2l = (l == 0) ? R20 : R2 + (size_t)(l - 1) * HD * HD;
    const float* Wml = (l == 0) ? Wm0 : Wm + (size_t)(l - 1) * HD * HD;
    const float* Wsl = (l == 0) ? Ws0 : Ws + (size_t)(l - 1) * HD * HD;
    gemm(basis, R1l, nullptr, t1, E, HD, NBAS, 1);        // relu(basis @ R1)
    gemm(t1,    R2l, nullptr, we, E, HD, HD,   0);        // @ R2
    gemm(Ah,    Wml, nullptr, hm, N, HD, Kh,   0);        // message projection (per-node)
    gemm(Ah,    Wsl, nullptr, hs, N, HD, Kh,   0);        // self path
    fillf(agg, 0.0f, (size_t)N * HD);
    msg_scatter_kernel<<<(unsigned)(((size_t)E * HD + 255) / 256), 256, 0, stream>>>(
        hm, ea, we, src, dst, agg, E);
    combine_kernel<<<(unsigned)(((size_t)N * HD + 255) / 256), 256, 0, stream>>>(
        hs, agg, hbuf, N, l < LAYERS - 1);
  }

  // TransformerConv
  gemm(hbuf, Wq, nullptr, q,  N, HEADS * HD, HD, 0);
  gemm(hbuf, Wk, nullptr, kb, N, HEADS * HD, HD, 0);
  gemm(hbuf, Wv, nullptr, v,  N, HEADS * HD, HD, 0);
  fillf(smax, -__builtin_huge_valf(), (size_t)N * HEADS);
  fillf(den, 0.0f, (size_t)N * HEADS);
  score_kernel<<<(unsigned)(((size_t)E * HEADS + 255) / 256), 256, 0, stream>>>(
      q, kb, src, dst, sc, smax, E);
  expden_kernel<<<(unsigned)(((size_t)E * HEADS + 255) / 256), 256, 0, stream>>>(
      sc, smax, dst, den, E);
  fillf(agg, 0.0f, (size_t)N * HD);
  attn_agg_kernel<<<(unsigned)(((size_t)E * HD + 255) / 256), 256, 0, stream>>>(
      sc, den, v, src, dst, agg, E);
  gemm(hbuf, Wskip, b_skip, hm, N, HD, HD, 0);            // hskip
  xlocal_kernel<<<(N + 255) / 256, 256, 0, stream>>>(agg, hm, hbuf, g_ln_l, b_ln_l, xc, N);

  // global attention pooling + global LN
  fillf(num, 0.0f, (size_t)GRAPHS * HD);
  fillf(cnt, 0.0f, (size_t)GRAPHS);
  ga_kernel<<<(N + 255) / 256, 256, 0, stream>>>(hbuf, W_ga, b_ga, batch, num, cnt, N);
  xglobal_kernel<<<(N + 255) / 256, 256, 0, stream>>>(hbuf, num, cnt, batch, g_ln_g, b_ln_g, xc, N);

  // projection head + classifier; outputs: [out(N) | proj(N*128)]
  float* out  = (float*)d_out;
  float* proj = out + N;
  gemm(xc,   P1, nullptr, thid, N, HD,  3 * HD, 1);
  gemm(thid, P2, nullptr, proj, N, 128, HD,     0);
  classifier_kernel<<<(N + 255) / 256, 256, 0, stream>>>(proj, Wc, bc, out, N);
}
